// GraphAttentionLayer_10617159156321
// MI455X (gfx1250) — compile-verified, matrix operand order checked by reading, not run on hardware
//
#include <hip/hip_runtime.h>

// GAT layer, MI455X (gfx1250, wave32, WMMA).
// Phase1: HW = H@W (f32, LDS-tiled) -> HWf16 + attn1/attn2 (f32).
// Phase2: per-row softmax stats (max, 1/sum) over masked leaky scores; one wave/row,
//         both k share each A read; attn2 staged in LDS.
// Phase3: fused attn^T @ HW with v_wmma_f32_16x16x32_f16; probabilities recomputed
//         on the fly into LDS f16 tiles (paired b32 stores); NT loads of A (read-once
//         per phase), prefetch of next chunk; bias+ReLU+transpose NT epilogue.

#define NN 8192
#define FF 512
#define KK 2
#define OO 64

typedef __attribute__((ext_vector_type(16))) _Float16 v16h;
typedef __attribute__((ext_vector_type(8)))  _Float16 v8h;
typedef __attribute__((ext_vector_type(2)))  _Float16 v2h;
typedef __attribute__((ext_vector_type(8)))  float    v8f;

constexpr float FMINV   = -3.402823466e38f;   // float32 min
constexpr float MASKVAL = FMINV * 0.2f;       // leaky(FMIN) -- exp underflows to 0

// ---------------------------------------------------------------------------
// Phase 1: HW[k,n,:] in f32 -> HWf16, attn1[k,n], attn2[k,n]
// block = 256 threads, 32 rows n per block; thread owns (n_local, 16 ko-slots)
// ---------------------------------------------------------------------------
__global__ __launch_bounds__(256) void gat_phase1(
    const float* __restrict__ H, const float* __restrict__ W,
    const float* __restrict__ a1, const float* __restrict__ a2,
    float* __restrict__ attn1, float* __restrict__ attn2,
    _Float16* __restrict__ HWh)
{
  __shared__ float Hs[32][64];    // 8 KB
  __shared__ float Ws[64][128];   // 32 KB
  const int t  = threadIdx.x;
  const int n0 = blockIdx.x * 32;
  const int nl     = t >> 3;        // 0..31
  const int kobase = (t & 7) * 16;  // 0..112 step 16

  float acc[16];
#pragma unroll
  for (int c = 0; c < 16; ++c) acc[c] = 0.f;

  for (int fc = 0; fc < FF / 64; ++fc) {
    const int f0 = fc * 64;
    __syncthreads();            // WAR guard vs previous chunk's reads
#pragma unroll
    for (int r = 0; r < 8; ++r) {       // 32x64 H tile
      int e = t + r * 256;
      Hs[e >> 6][e & 63] = H[(size_t)(n0 + (e >> 6)) * FF + f0 + (e & 63)];
    }
    for (int r = 0; r < 32; ++r) {      // 64x128 W tile (ko = k*64+o)
      int e = t + r * 256;
      int f = e >> 7, ko = e & 127;
      Ws[f][ko] = W[((size_t)(ko >> 6) * FF + f0 + f) * OO + (ko & 63)];
    }
    __syncthreads();
    for (int f = 0; f < 64; ++f) {
      float h = Hs[nl][f];
#pragma unroll
      for (int c = 0; c < 16; ++c) acc[c] = fmaf(h, Ws[f][kobase + c], acc[c]);
    }
  }

  const int k  = kobase >> 6;
  const int ob = kobase & 63;
  float p1 = 0.f, p2 = 0.f;
#pragma unroll
  for (int c = 0; c < 16; ++c) {
    p1 = fmaf(acc[c], a1[k * OO + ob + c], p1);
    p2 = fmaf(acc[c], a2[k * OO + ob + c], p2);
  }
  // reduce across the 4 threads sharing (n_local, k): lane-xor 1 and 2
  p1 += __shfl_xor(p1, 1); p1 += __shfl_xor(p1, 2);
  p2 += __shfl_xor(p2, 1); p2 += __shfl_xor(p2, 2);

  const int n = n0 + nl;
  if ((t & 3) == 0) { attn1[k * NN + n] = p1; attn2[k * NN + n] = p2; }
#pragma unroll
  for (int c = 0; c < 16; ++c)
    HWh[((size_t)k * NN + n) * OO + ob + c] = (_Float16)acc[c];
}

// ---------------------------------------------------------------------------
// Phase 2: per-row max and 1/sum of exp for both k. One wave per row j.
// ---------------------------------------------------------------------------
__global__ __launch_bounds__(256) void gat_phase2(
    const float* __restrict__ A, const int* __restrict__ idx,
    const float* __restrict__ attn1, const float* __restrict__ attn2,
    float* __restrict__ mrow, float* __restrict__ rlrow)
{
  __shared__ float at2[KK][NN];   // 64 KB
  const int t = threadIdx.x;
  for (int e = t; e < KK * NN; e += 256) at2[e >> 13][e & (NN - 1)] = attn2[e];
  __syncthreads();

  const int wave = t >> 5, lane = t & 31;
  const int j = blockIdx.x * 8 + wave;
  const float a10 = attn1[j], a11 = attn1[NN + j];
  const size_t rowoff = (size_t)j * NN;

  float m0 = MASKVAL, m1 = MASKVAL;
  for (int i = lane; i < NN; i += 32) {
    float a  = A[rowoff + idx[i]];
    float v0 = a10 + at2[0][i]; v0 = (v0 >= 0.f) ? v0 : 0.2f * v0; v0 = (a != 0.f) ? v0 : MASKVAL;
    float v1 = a11 + at2[1][i]; v1 = (v1 >= 0.f) ? v1 : 0.2f * v1; v1 = (a != 0.f) ? v1 : MASKVAL;
    m0 = fmaxf(m0, v0); m1 = fmaxf(m1, v1);
  }
  for (int off = 16; off; off >>= 1) {
    m0 = fmaxf(m0, __shfl_xor(m0, off));
    m1 = fmaxf(m1, __shfl_xor(m1, off));
  }

  float l0 = 0.f, l1 = 0.f;
  for (int i = lane; i < NN; i += 32) {
    float a  = A[rowoff + idx[i]];
    float v0 = a10 + at2[0][i]; v0 = (v0 >= 0.f) ? v0 : 0.2f * v0; v0 = (a != 0.f) ? v0 : MASKVAL;
    float v1 = a11 + at2[1][i]; v1 = (v1 >= 0.f) ? v1 : 0.2f * v1; v1 = (a != 0.f) ? v1 : MASKVAL;
    l0 += __expf(v0 - m0);      // masked -> exp(huge negative) == 0 exactly
    l1 += __expf(v1 - m1);
  }
  for (int off = 16; off; off >>= 1) {
    l0 += __shfl_xor(l0, off);
    l1 += __shfl_xor(l1, off);
  }
  if (lane == 0) {
    mrow[j] = m0;  mrow[NN + j] = m1;
    rlrow[j] = 1.f / l0;  rlrow[NN + j] = 1.f / l1;
  }
}

// ---------------------------------------------------------------------------
// Phase 3: res[k,i,o] = sum_j p[k,j,i] * HW[k,j,o] via WMMA f16->f32.
// Block: 128-wide i tile, both k, all 64 o. 8 waves: wave w -> (k=w>>2, o-slab=(w&3)*16).
// ---------------------------------------------------------------------------
__global__ __launch_bounds__(256) void gat_phase3(
    const float* __restrict__ A, const int* __restrict__ idx,
    const float* __restrict__ attn1, const float* __restrict__ attn2,
    const float* __restrict__ mrow, const float* __restrict__ rlrow,
    const _Float16* __restrict__ HWh, const float* __restrict__ bvec,
    float* __restrict__ out)
{
  __shared__ _Float16 pT[KK][128][32];   // P^T tile (i-major), 16 KB
  __shared__ _Float16 hwT[KK][OO][32];   // HW^T tile (o-major), 8 KB
  __shared__ float at2s[KK][128];
  __shared__ float jm_a1[KK][32], jm_m[KK][32], jm_rl[KK][32];
  __shared__ int   ixs[128];

  const int t     = threadIdx.x;
  const int lane  = t & 31;
  const int w     = t >> 5;
  const int kw    = w >> 2;           // 0..1
  const int otile = (w & 3) * 16;     // 0..48
  const int i0    = blockIdx.x * 128;

  { int k = t >> 7, il = t & 127; at2s[k][il] = attn2[k * NN + i0 + il]; }
  if (t < 128) ixs[t] = idx[i0 + t];

  v8f acc[8] = {};

  const int nn2 = lane & 15;          // B col / C col (o), A row (i) within tile
  const int lh  = lane >> 4;          // half-select for K packing
  const float bv = bvec[kw * OO + otile + nn2];

  union FragU { v16h f; v8h h[2]; };

  for (int nc = 0; nc < NN / 32; ++nc) {
    const int j0 = nc * 32;

    // prefetch next chunk's A tile (16 KB): 256 threads x one 64B segment
    if (nc + 1 < NN / 32) {
      const float* pfA = A + (size_t)(j0 + 32 + (t >> 3)) * NN + i0 + (t & 7) * 16;
      __builtin_prefetch(pfA, 0, 1);
    }

    __syncthreads();                  // WAR: previous chunk's frag reads done
    if (t < 64) {
      int k = t >> 5, jl = t & 31, jg = k * NN + j0 + jl;
      jm_a1[k][jl] = attn1[jg];
      jm_m [k][jl] = mrow [jg];
      jm_rl[k][jl] = rlrow[jg];
    }
#pragma unroll
    for (int r = 0; r < 8; ++r) {     // hwT fill: 2k x 16 j-pairs x 64 o, b32 stores
      int e   = t + r * 256;          // 0..2047
      int k   = e >> 10;
      int rem = e & 1023;
      int jl2 = rem >> 6, o = rem & 63;
      _Float16 h0 = HWh[((size_t)k * NN + j0 + jl2 * 2    ) * OO + o];
      _Float16 h1 = HWh[((size_t)k * NN + j0 + jl2 * 2 + 1) * OO + o];
      v2h pk = { h0, h1 };
      *(v2h*)&hwT[k][o][jl2 * 2] = pk;
    }
    __syncthreads();                  // jm_* visible
#pragma unroll
    for (int r = 0; r < 8; ++r) {     // pT fill: 16 j-pairs x 128 i, both k per A read
      int e   = t + r * 256;          // 0..2047
      int jl2 = e >> 7, il = e & 127;
      int jlA = jl2 * 2, jlB = jlA + 1;
      const int col = ixs[il];
      float aA = __builtin_nontemporal_load(&A[(size_t)(j0 + jlA) * NN + col]);
      float aB = __builtin_nontemporal_load(&A[(size_t)(j0 + jlB) * NN + col]);
#pragma unroll
      for (int k = 0; k < KK; ++k) {
        float t2 = at2s[k][il];
        float vA = jm_a1[k][jlA] + t2;
        vA = (vA >= 0.f) ? vA : 0.2f * vA;
        vA = (aA != 0.f) ? vA : MASKVAL;
        float vB = jm_a1[k][jlB] + t2;
        vB = (vB >= 0.f) ? vB : 0.2f * vB;
        vB = (aB != 0.f) ? vB : MASKVAL;
        float pA = __expf(vA - jm_m[k][jlA]) * jm_rl[k][jlA];
        float pB = __expf(vB - jm_m[k][jlB]) * jm_rl[k][jlB];
        v2h pk = { (_Float16)pA, (_Float16)pB };
        *(v2h*)&pT[k][il][jlA] = pk;
      }
    }
    __syncthreads();                  // tiles ready

    // B fragment: K(j)=32 x N(o)=16, ISA layout: lane<16 K 0..7/16..23, lane>=16 K 8..15/24..31
    FragU bf;
    bf.h[0] = *(const v8h*)&hwT[kw][otile + nn2][lh * 8];
    bf.h[1] = *(const v8h*)&hwT[kw][otile + nn2][16 + lh * 8];

#pragma unroll
    for (int s = 0; s < 8; ++s) {
      FragU af;                       // A fragment: M(i)=16 x K(j)=32
      af.h[0] = *(const v8h*)&pT[kw][s * 16 + nn2][lh * 8];
      af.h[1] = *(const v8h*)&pT[kw][s * 16 + nn2][16 + lh * 8];
      acc[s] = __builtin_amdgcn_wmma_f32_16x16x32_f16(
          false, af.f, false, bf.f, (short)0, acc[s], false, false);
    }
  }

  // Epilogue: bias + ReLU + transpose to out[n, k*O+o], non-temporal (write-once)
  const int mbase = lh * 8;
#pragma unroll
  for (int s = 0; s < 8; ++s) {
#pragma unroll
    for (int r = 0; r < 8; ++r) {
      int i = i0 + s * 16 + mbase + r;
      float val = fmaxf(acc[s][r] + bv, 0.f);
      __builtin_nontemporal_store(val, &out[(size_t)i * (KK * OO) + kw * OO + otile + nn2]);
    }
  }
}

// ---------------------------------------------------------------------------
extern "C" void kernel_launch(void* const* d_in, const int* in_sizes, int n_in,
                              void* d_out, int out_size, void* d_ws, size_t ws_size,
                              hipStream_t stream)
{
  const float* H   = (const float*)d_in[0];
  const float* A   = (const float*)d_in[1];
  const int*   idx = (const int*)  d_in[2];
  const float* W   = (const float*)d_in[3];
  const float* b   = (const float*)d_in[4];
  const float* a1  = (const float*)d_in[5];
  const float* a2  = (const float*)d_in[6];
  float* out = (float*)d_out;

  char* ws = (char*)d_ws;
  float*    attn1 = (float*)(ws);
  float*    attn2 = (float*)(ws + (size_t)64 * 1024);
  float*    mrow  = (float*)(ws + (size_t)128 * 1024);
  float*    rlrow = (float*)(ws + (size_t)192 * 1024);
  _Float16* HWh   = (_Float16*)(ws + (size_t)256 * 1024);   // 2 MB

  gat_phase1<<<dim3(NN / 32),  dim3(256), 0, stream>>>(H, W, a1, a2, attn1, attn2, HWh);
  gat_phase2<<<dim3(NN / 8),   dim3(256), 0, stream>>>(A, idx, attn1, attn2, mrow, rlrow);
  gat_phase3<<<dim3(NN / 128), dim3(256), 0, stream>>>(A, idx, attn1, attn2, mrow, rlrow,
                                                       HWh, b, out);
}
